// SparseTransformer_47425028883013
// MI455X (gfx1250) — matrix operand order repaired and checked by reading, not compiled, not run on hardware
//
#include <hip/hip_runtime.h>
#include <hip/hip_bf16.h>
#include <math.h>

// ---------------------------------------------------------------------------
// SparseTransformer forward for gfx1250 (MI455X).  bf16 WMMA everywhere heavy.
// Round 3: fix async-LDS builtin pointer types (v4i addrspace pointers).
// ---------------------------------------------------------------------------

typedef __bf16 bf16_t;
typedef bf16_t v16bf __attribute__((ext_vector_type(16)));
typedef bf16_t v8bf  __attribute__((ext_vector_type(8)));
typedef float  v8f   __attribute__((ext_vector_type(8)));

#define NTOK   16384
#define CDIM   512
#define NHEAD  8
#define DHEAD  64
#define WINSZ  1024
#define NLAYER 8
#define CFFD   2048
#define CINP   8
#define COUTP  8
#define FREQD  256

#define MODE_BF16  0
#define MODE_GELU  1
#define MODE_GATED 2

// ---- CDNA5 async global->LDS copy (GLOBAL_LOAD_ASYNC_TO_LDS_B128) ---------
#if defined(__AMDGCN__) && __has_builtin(__builtin_amdgcn_global_load_async_to_lds_b128)
#define HAVE_ASYNC_LDS 1
#else
#define HAVE_ASYNC_LDS 0
#endif

#if HAVE_ASYNC_LDS
typedef int v4i_gcc __attribute__((vector_size(16)));
typedef __attribute__((address_space(1))) v4i_gcc* g_v4ip;
typedef __attribute__((address_space(3))) v4i_gcc* l_v4ip;
static __device__ inline void async_copy16(void* lds_dst, const void* gsrc) {
    __builtin_amdgcn_global_load_async_to_lds_b128(
        (g_v4ip)gsrc, (l_v4ip)lds_dst, 0, 0);
}
static __device__ inline void async_wait0() {
#if __has_builtin(__builtin_amdgcn_s_wait_asynccnt)
    __builtin_amdgcn_s_wait_asynccnt(0);
#else
    asm volatile("s_wait_asynccnt 0x0" ::: "memory");
#endif
}
#else
static __device__ inline void async_wait0() {}
#endif

static __device__ inline v8f wmma_bf16(v16bf a, v16bf b, v8f c) {
    return __builtin_amdgcn_wmma_f32_16x16x32_bf16(
        /*neg_a=*/false, a, /*neg_b=*/false, b,
        /*c_mod=*/(short)0, c, /*reuse_a=*/false, /*reuse_b=*/false);
}

static __device__ inline float gelu_t(float x) {
    float x3 = x * x * x;
    return 0.5f * x * (1.f + tanhf(0.79788456f * (x + 0.044715f * x3)));
}
static __device__ inline float siluf(float x) { return x / (1.f + __expf(-x)); }

static __device__ inline float redmax16(float v) {
#pragma unroll
    for (int m = 1; m <= 8; m <<= 1) v = fmaxf(v, __shfl_xor(v, m, 32));
    return v;
}
static __device__ inline float redsum16(float v) {
#pragma unroll
    for (int m = 1; m <= 8; m <<= 1) v += __shfl_xor(v, m, 32);
    return v;
}

// ---------------------------------------------------------------------------
// h = feats @ in_w.T + in_b + pos_embed(coords)
// ---------------------------------------------------------------------------
__global__ void embed_kernel(const float* __restrict__ feats,
                             const int* __restrict__ coords,
                             const float* __restrict__ in_w,
                             const float* __restrict__ in_b,
                             float* __restrict__ h) {
    int idx = blockIdx.x * 256 + threadIdx.x;      // N*C elements
    int i = idx >> 9, c = idx & 511;
    float acc = in_b[c];
#pragma unroll
    for (int k = 0; k < CINP; k++) acc += feats[i * CINP + k] * in_w[c * CINP + k];
    if (c < 510) {                                  // 3 coords * 2 * 85 = 510
        int a = c / 170, wv = c % 170;
        int kk = (wv < 85) ? wv : wv - 85;
        float f = __expf(-logf(10000.f) * (float)kk / 85.f);
        float o = (float)coords[i * 3 + a] * f;
        acc += (wv < 85) ? __sinf(o) : __cosf(o);
    }
    h[idx] = acc;
}

// ---------------------------------------------------------------------------
// silu_c = silu( silu(emb @ w1.T + b1) @ w2.T + b2 )  (outer silu folded in:
// only silu(c) is ever consumed, by adaLN)
// ---------------------------------------------------------------------------
__global__ void t_embed_kernel(const float* __restrict__ t,
                               const float* __restrict__ w1, const float* __restrict__ b1,
                               const float* __restrict__ w2, const float* __restrict__ b2,
                               float* __restrict__ silu_c) {
    __shared__ float l1[CDIM];
    float tv = t[0];
    for (int o = threadIdx.x; o < CDIM; o += 256) {
        float acc = b1[o];
        for (int k = 0; k < FREQD; k++) {
            int kk = (k < 128) ? k : k - 128;
            float f = __expf(-logf(10000.f) * (float)kk / 128.f);
            float arg = tv * f;
            float e = (k < 128) ? __cosf(arg) : __sinf(arg);
            acc += e * w1[o * FREQD + k];
        }
        l1[o] = siluf(acc);
    }
    __syncthreads();
    for (int o = threadIdx.x; o < CDIM; o += 256) {
        float acc = b2[o];
        for (int k = 0; k < CDIM; k++) acc += l1[k] * w2[o * CDIM + k];
        silu_c[o] = siluf(acc);
    }
}

// mod = silu_c @ ada_w.T + ada_b   (1 x 3072)
__global__ void ada_kernel(const float* __restrict__ silu_c,
                           const float* __restrict__ aw, const float* __restrict__ ab,
                           float* __restrict__ mod) {
    __shared__ float cs[CDIM];
    for (int k = threadIdx.x; k < CDIM; k += 256) cs[k] = silu_c[k];
    __syncthreads();
    int o = blockIdx.x * 256 + threadIdx.x;        // 12 blocks -> 3072
    float acc = ab[o];
    for (int k = 0; k < CDIM; k++) acc += cs[k] * aw[(size_t)o * CDIM + k];
    mod[o] = acc;
}

// hn = LN(h) * (1+sc) + sm     -> bf16, one wave per row
__global__ void ln_mod_kernel(const float* __restrict__ h,
                              const float* __restrict__ sm, const float* __restrict__ sc,
                              bf16_t* __restrict__ hn) {
    int wave = threadIdx.x >> 5, lane = threadIdx.x & 31;
    int row = blockIdx.x * 8 + wave;
    const float* hr = h + (size_t)row * CDIM;
    float x[16], s = 0.f, s2 = 0.f;
#pragma unroll
    for (int k = 0; k < 16; k++) {
        x[k] = hr[lane + k * 32];
        s += x[k]; s2 += x[k] * x[k];
    }
#pragma unroll
    for (int m = 1; m <= 16; m <<= 1) { s += __shfl_xor(s, m, 32); s2 += __shfl_xor(s2, m, 32); }
    float mean = s * (1.f / 512.f);
    float var  = s2 * (1.f / 512.f) - mean * mean;
    float inv  = rsqrtf(var + 1e-6f);
#pragma unroll
    for (int k = 0; k < 16; k++) {
        int c = lane + k * 32;
        float v = (x[k] - mean) * inv * (1.f + sc[c]) + sm[c];
        hn[(size_t)row * CDIM + c] = (bf16_t)v;
    }
}

__global__ void cvt_bf16_kernel(const float* __restrict__ src, bf16_t* __restrict__ dst, int n) {
    int i = blockIdx.x * 256 + threadIdx.x;
    if (i < n) dst[i] = (bf16_t)src[i];
}

// ---------------------------------------------------------------------------
// Generic bf16 WMMA GEMM:  Y(N x M) = X(N x K, bf16) @ W(M x K, bf16)^T + b
// Block 64x128, 8 waves (2x4), 32x32 per wave, K-step 32.
// Double-buffered LDS; tiles streamed with GLOBAL_LOAD_ASYNC_TO_LDS_B128
// (fallback: register-staged copies).  One barrier per K-step.
// Epilogues: MODE_BF16 (qkv), MODE_GELU (fc1), MODE_GATED (proj/fc2 residual)
// ---------------------------------------------------------------------------
__global__ __launch_bounds__(256) void gemm_bf16_kernel(
        const bf16_t* __restrict__ X, const bf16_t* __restrict__ W,
        const float* __restrict__ bias, float* __restrict__ hres,
        const float* __restrict__ gate, bf16_t* __restrict__ outb,
        int M, int K, int mode) {
    __shared__ bf16_t Xs[2][64 * 40];               // row stride 40 (80B, 16B-aligned)
    __shared__ bf16_t Ws[2][128 * 40];
    const int tid  = threadIdx.x;
    const int lane = tid & 31, wave = tid >> 5;
    const int wm = wave >> 2, wn = wave & 3;
    const int half = lane >> 4, ln = lane & 15;
    const int rowBlk = blockIdx.x * 64;
    const int colBlk = blockIdx.y * 128;

    v8f acc[2][2];
#pragma unroll
    for (int mi = 0; mi < 2; mi++)
#pragma unroll
        for (int ni = 0; ni < 2; ni++)
#pragma unroll
            for (int j = 0; j < 8; j++) acc[mi][ni][j] = 0.f;

    // cooperative tile copy: thread -> one 16B chunk of X, two of W
    const int xr = tid >> 2, xc = (tid & 3) * 8;    // X row 0..63, W rows xr / xr+64
    const int wr1 = xr + 64;
    const bf16_t* gX  = &X[(size_t)(rowBlk + xr) * K + xc];
    const bf16_t* gW0 = &W[(size_t)(colBlk + xr) * K + xc];
    const bf16_t* gW1 = &W[(size_t)(colBlk + wr1) * K + xc];
    const int lofX  = xr * 40 + xc;
    const int lofW0 = xr * 40 + xc;
    const int lofW1 = wr1 * 40 + xc;

    const int nk = K >> 5;

#if HAVE_ASYNC_LDS
    async_copy16(&Xs[0][lofX],  gX);
    async_copy16(&Ws[0][lofW0], gW0);
    async_copy16(&Ws[0][lofW1], gW1);
#else
    {
        v8bf rX  = *(const v8bf*)gX;
        v8bf rW0 = *(const v8bf*)gW0;
        v8bf rW1 = *(const v8bf*)gW1;
        *(v8bf*)&Xs[0][lofX]  = rX;
        *(v8bf*)&Ws[0][lofW0] = rW0;
        *(v8bf*)&Ws[0][lofW1] = rW1;
    }
#endif

    for (int kt = 0; kt < nk; kt++) {
        const int cur = kt & 1;
        const int nxt = cur ^ 1;
        const int k1 = (kt + 1) << 5;
#if HAVE_ASYNC_LDS
        async_wait0();                              // tile kt resident in LDS
        __syncthreads();
        if (kt + 1 < nk) {                          // stream tile kt+1 under compute
            async_copy16(&Xs[nxt][lofX],  gX  + k1);
            async_copy16(&Ws[nxt][lofW0], gW0 + k1);
            async_copy16(&Ws[nxt][lofW1], gW1 + k1);
        }
#else
        __syncthreads();
        v8bf nX = {}, nW0 = {}, nW1 = {};
        if (kt + 1 < nk) {                          // issue all loads up front
            nX  = *(const v8bf*)(gX  + k1);
            nW0 = *(const v8bf*)(gW0 + k1);
            nW1 = *(const v8bf*)(gW1 + k1);
        }
#endif
        v16bf a[2], b[2];
#pragma unroll
        for (int mi = 0; mi < 2; mi++) {            // A frag: row=ln, kb=half*8
            const bf16_t* p = &Xs[cur][(wm * 32 + mi * 16 + ln) * 40 + half * 8];
            *(v8bf*)&a[mi]         = *(const v8bf*)p;         // k = kb..kb+7
            *(((v8bf*)&a[mi]) + 1) = *(const v8bf*)(p + 16);  // k = kb+16..kb+23
        }
#pragma unroll
        for (int ni = 0; ni < 2; ni++) {            // B frag: n=ln, kb=half*8 (same k order as A)
            const bf16_t* p = &Ws[cur][(wn * 32 + ni * 16 + ln) * 40 + half * 8];
            *(v8bf*)&b[ni]         = *(const v8bf*)p;
            *(((v8bf*)&b[ni]) + 1) = *(const v8bf*)(p + 16);
        }
#pragma unroll
        for (int mi = 0; mi < 2; mi++)
#pragma unroll
            for (int ni = 0; ni < 2; ni++)
                acc[mi][ni] = wmma_bf16(a[mi], b[ni], acc[mi][ni]);
#if !HAVE_ASYNC_LDS
        if (kt + 1 < nk) {
            *(v8bf*)&Xs[nxt][lofX]  = nX;
            *(v8bf*)&Ws[nxt][lofW0] = nW0;
            *(v8bf*)&Ws[nxt][lofW1] = nW1;
        }
#endif
    }

#pragma unroll
    for (int mi = 0; mi < 2; mi++)
#pragma unroll
        for (int ni = 0; ni < 2; ni++) {
            int colm = colBlk + wn * 32 + ni * 16 + ln;
            float bv = bias[colm];
#pragma unroll
            for (int j = 0; j < 8; j++) {
                int r = rowBlk + wm * 32 + mi * 16 + j + half * 8;
                float v = acc[mi][ni][j] + bv;
                if (mode == MODE_GATED)       hres[(size_t)r * M + colm] += gate[colm] * v;
                else if (mode == MODE_GELU)   outb[(size_t)r * M + colm] = (bf16_t)gelu_t(v);
                else                          outb[(size_t)r * M + colm] = (bf16_t)v;
            }
        }
}

// ---------------------------------------------------------------------------
// Flash attention over one (window, head, 128-row tile). qkv bf16 N x 1536.
// Online softmax; S and P*V via bf16 WMMA.  K tile staged in LDS (async) --
// all 8 waves share it; V tile staged transposed for contiguous B-frags.
// ---------------------------------------------------------------------------
__global__ __launch_bounds__(256) void attn_kernel(const bf16_t* __restrict__ qkv,
                                                   bf16_t* __restrict__ attno,
                                                   int shift) {
    __shared__ bf16_t Kls[32 * 72];                 // [key][dim], stride 72 (144B)
    __shared__ bf16_t Vt[64 * 40];                  // [dim][key], stride 40
    __shared__ bf16_t Pls[8][16 * 40];              // per-wave P patch
    const int tid  = threadIdx.x;
    const int lane = tid & 31, wave = tid >> 5;
    const int half = lane >> 4, ln = lane & 15;
    const int bx = blockIdx.x;
    const int w  = bx >> 6;
    const int hh = (bx >> 3) & 7;
    const int mt = bx & 7;
    const int rowbase = mt * 128 + wave * 16;       // rolled space, within window
    const int msk = NTOK - 1;
    const int ldq = 3 * CDIM;

    // Q fragments (rows fixed for whole kernel)
    v16bf aq[2];
    {
        int qrow = (w * WINSZ + rowbase + ln + shift) & msk;
        const bf16_t* qp = qkv + (size_t)qrow * ldq + hh * DHEAD;
        int kb = half * 8;
#pragma unroll
        for (int d0 = 0; d0 < 2; d0++) {
            const bf16_t* p = qp + d0 * 32 + kb;
            *(v8bf*)&aq[d0]         = *(const v8bf*)p;
            *(((v8bf*)&aq[d0]) + 1) = *(const v8bf*)(p + 16);
        }
    }

    v8f o[4];
#pragma unroll
    for (int ni = 0; ni < 4; ni++)
#pragma unroll
        for (int j = 0; j < 8; j++) o[ni][j] = 0.f;
    float mrow[8], lrow[8];
#pragma unroll
    for (int j = 0; j < 8; j++) { mrow[j] = -1e30f; lrow[j] = 0.f; }
    const float scale = 0.125f;                     // DH^-0.5

    // cooperative copy geometry: K tile 32x64 -> thread = 16B chunk
    const int kkey = tid >> 3, kdb = (tid & 7) * 8;
    // V tile 32x64, transposed store: thread loads row kk, 8 dims
    const int vkk = tid & 31, vdb = (tid >> 5) * 8;

    for (int j0 = 0; j0 < WINSZ; j0 += 32) {
        __syncthreads();
        {   // K tile -> LDS (shared by all 8 waves; was 8x redundant global)
            int krow = (w * WINSZ + j0 + kkey + shift) & msk;
            const bf16_t* kp = qkv + (size_t)krow * ldq + CDIM + hh * DHEAD + kdb;
#if HAVE_ASYNC_LDS
            async_copy16(&Kls[kkey * 72 + kdb], kp);
#else
            *(v8bf*)&Kls[kkey * 72 + kdb] = *(const v8bf*)kp;
#endif
        }
        {   // V tile transposed: Vt[d][k]
            int vrow = (w * WINSZ + j0 + vkk + shift) & msk;
            const bf16_t* vp = qkv + (size_t)vrow * ldq + 2 * CDIM + hh * DHEAD + vdb;
            v8bf vv = *(const v8bf*)vp;
#pragma unroll
            for (int q = 0; q < 8; q++) Vt[(vdb + q) * 40 + vkk] = vv[q];
        }
        async_wait0();
        __syncthreads();

        // S tiles (2 x 16 keys), B-frags from LDS K tile (contiguous in d)
        v8f s[2];
#pragma unroll
        for (int jt = 0; jt < 2; jt++) {
            const bf16_t* kb0 = &Kls[(jt * 16 + ln) * 72 + half * 8];
            v16bf bk0, bk1;
            *(v8bf*)&bk0         = *(const v8bf*)(kb0);
            *(((v8bf*)&bk0) + 1) = *(const v8bf*)(kb0 + 16);
            *(v8bf*)&bk1         = *(const v8bf*)(kb0 + 32);
            *(((v8bf*)&bk1) + 1) = *(const v8bf*)(kb0 + 48);
            v8f st;
#pragma unroll
            for (int j = 0; j < 8; j++) st[j] = 0.f;
            st = wmma_bf16(aq[0], bk0, st);
            st = wmma_bf16(aq[1], bk1, st);
#pragma unroll
            for (int j = 0; j < 8; j++) st[j] *= scale;
            s[jt] = st;
        }

        // online softmax, per row slot j (+ half*8)
#pragma unroll
        for (int j = 0; j < 8; j++) {
            float rmax = redmax16(fmaxf(s[0][j], s[1][j]));
            float nm = fmaxf(mrow[j], rmax);
            float corr = __expf(mrow[j] - nm);
            float p0 = __expf(s[0][j] - nm);
            float p1 = __expf(s[1][j] - nm);
            float rs = redsum16(p0 + p1);
            lrow[j] = lrow[j] * corr + rs;
            mrow[j] = nm;
#pragma unroll
            for (int ni = 0; ni < 4; ni++) o[ni][j] *= corr;
            int prow = j + half * 8;
            Pls[wave][prow * 40 + ln]      = (bf16_t)p0;
            Pls[wave][prow * 40 + 16 + ln] = (bf16_t)p1;
        }

        // P as A-frag (wave-private LDS round trip; DS ops in-order per wave)
        v16bf ap;
        {
            const bf16_t* pb = &Pls[wave][ln * 40 + half * 8];
            *(v8bf*)&ap         = *(const v8bf*)pb;
            *(((v8bf*)&ap) + 1) = *(const v8bf*)(pb + 16);
        }
        // O += P @ V, B-frags contiguous from transposed Vt
#pragma unroll
        for (int ni = 0; ni < 4; ni++) {
            v16bf bv;
            const bf16_t* vb = &Vt[(ni * 16 + ln) * 40 + half * 8];
            *(v8bf*)&bv         = *(const v8bf*)vb;
            *(((v8bf*)&bv) + 1) = *(const v8bf*)(vb + 16);
            o[ni] = wmma_bf16(ap, bv, o[ni]);
        }
    }

    // write normalized output (roll back by +shift == same index mapping)
#pragma unroll
    for (int ni = 0; ni < 4; ni++)
#pragma unroll
        for (int j = 0; j < 8; j++) {
            int rr = rowbase + j + half * 8;
            int grow = (w * WINSZ + rr + shift) & msk;
            float val = o[ni][j] / lrow[j];
            attno[(size_t)grow * CDIM + hh * DHEAD + ni * 16 + ln] = (bf16_t)val;
        }
}

// y = h @ out_w.T + out_b   (N x 8)
__global__ void out_kernel(const float* __restrict__ h,
                           const float* __restrict__ ow, const float* __restrict__ ob,
                           float* __restrict__ y) {
    int idx = blockIdx.x * 256 + threadIdx.x;       // N*8
    int i = idx >> 3, o = idx & 7;
    float acc = ob[o];
    for (int k = 0; k < CDIM; k++) acc += h[(size_t)i * CDIM + k] * ow[o * CDIM + k];
    y[idx] = acc;
}

// ---------------------------------------------------------------------------
// Host side
// ---------------------------------------------------------------------------
extern "C" void kernel_launch(void* const* d_in, const int* in_sizes, int n_in,
                              void* d_out, int out_size, void* d_ws, size_t ws_size,
                              hipStream_t stream) {
    (void)in_sizes; (void)n_in; (void)out_size; (void)ws_size;
    const float* feats  = (const float*)d_in[0];
    const int*   coords = (const int*)  d_in[1];
    const float* tval   = (const float*)d_in[2];
    const float* in_w   = (const float*)d_in[3];
    const float* in_b   = (const float*)d_in[4];
    const float* t_w1   = (const float*)d_in[5];
    const float* t_b1   = (const float*)d_in[6];
    const float* t_w2   = (const float*)d_in[7];
    const float* t_b2   = (const float*)d_in[8];
    const float* qkv_w  = (const float*)d_in[9];
    const float* qkv_b  = (const float*)d_in[10];
    const float* proj_w = (const float*)d_in[11];
    const float* proj_b = (const float*)d_in[12];
    const float* ada_w  = (const float*)d_in[13];
    const float* ada_b  = (const float*)d_in[14];
    const float* fc1_w  = (const float*)d_in[15];
    const float* fc1_b  = (const float*)d_in[16];
    const float* fc2_w  = (const float*)d_in[17];
    const float* fc2_b  = (const float*)d_in[18];
    const float* out_w  = (const float*)d_in[19];
    const float* out_b  = (const float*)d_in[20];

    char* ws = (char*)d_ws;
    size_t off = 0;
    float*  silu_c = (float*)(ws + off);  off += 4096;
    float*  mod    = (float*)(ws + off);  off += 16384;
    float*  h      = (float*)(ws + off);  off += (size_t)NTOK * CDIM * 4;      // 33.5 MB
    bf16_t* hn     = (bf16_t*)(ws + off); off += (size_t)NTOK * CDIM * 2;      // 16.8 MB
    bf16_t* qkvb   = (bf16_t*)(ws + off); off += (size_t)NTOK * 3 * CDIM * 2;  // 50.3 MB
    bf16_t* attno  = (bf16_t*)(ws + off); off += (size_t)NTOK * CDIM * 2;      // 16.8 MB
    bf16_t* mlpb   = (bf16_t*)(ws + off); off += (size_t)NTOK * CFFD * 2;      // 67.1 MB
    bf16_t* wbuf   = (bf16_t*)(ws + off); off += (size_t)CFFD * CDIM * 2;      //  2.1 MB

    embed_kernel<<<(NTOK * CDIM) / 256, 256, 0, stream>>>(feats, coords, in_w, in_b, h);
    t_embed_kernel<<<1, 256, 0, stream>>>(tval, t_w1, t_b1, t_w2, t_b2, silu_c);

    for (int i = 0; i < NLAYER; i++) {
        int shift = (i & 1) ? (WINSZ / 2) : 0;
        ada_kernel<<<12, 256, 0, stream>>>(silu_c,
            ada_w + (size_t)i * 6 * CDIM * CDIM, ada_b + (size_t)i * 6 * CDIM, mod);

        // ---- attention branch ----
        ln_mod_kernel<<<NTOK / 8, 256, 0, stream>>>(h, mod + 0 * CDIM, mod + 1 * CDIM, hn);
        cvt_bf16_kernel<<<(3 * CDIM * CDIM + 255) / 256, 256, 0, stream>>>(
            qkv_w + (size_t)i * 3 * CDIM * CDIM, wbuf, 3 * CDIM * CDIM);
        gemm_bf16_kernel<<<dim3(NTOK / 64, (3 * CDIM) / 128), 256, 0, stream>>>(
            hn, wbuf, qkv_b + (size_t)i * 3 * CDIM, nullptr, nullptr, qkvb,
            3 * CDIM, CDIM, MODE_BF16);
        attn_kernel<<<(NTOK / WINSZ) * NHEAD * (WINSZ / 128), 256, 0, stream>>>(
            qkvb, attno, shift);
        cvt_bf16_kernel<<<(CDIM * CDIM + 255) / 256, 256, 0, stream>>>(
            proj_w + (size_t)i * CDIM * CDIM, wbuf, CDIM * CDIM);
        gemm_bf16_kernel<<<dim3(NTOK / 64, CDIM / 128), 256, 0, stream>>>(
            attno, wbuf, proj_b + (size_t)i * CDIM, h, mod + 2 * CDIM, nullptr,
            CDIM, CDIM, MODE_GATED);

        // ---- MLP branch ----
        ln_mod_kernel<<<NTOK / 8, 256, 0, stream>>>(h, mod + 3 * CDIM, mod + 4 * CDIM, hn);
        cvt_bf16_kernel<<<(CFFD * CDIM + 255) / 256, 256, 0, stream>>>(
            fc1_w + (size_t)i * CFFD * CDIM, wbuf, CFFD * CDIM);
        gemm_bf16_kernel<<<dim3(NTOK / 64, CFFD / 128), 256, 0, stream>>>(
            hn, wbuf, fc1_b + (size_t)i * CFFD, nullptr, nullptr, mlpb,
            CFFD, CDIM, MODE_GELU);
        cvt_bf16_kernel<<<(CDIM * CFFD + 255) / 256, 256, 0, stream>>>(
            fc2_w + (size_t)i * CDIM * CFFD, wbuf, CDIM * CFFD);
        gemm_bf16_kernel<<<dim3(NTOK / 64, CDIM / 128), 256, 0, stream>>>(
            mlpb, wbuf, fc2_b + (size_t)i * CDIM, h, mod + 5 * CDIM, nullptr,
            CDIM, CFFD, MODE_GATED);
    }

    out_kernel<<<(NTOK * COUTP) / 256, 256, 0, stream>>>(h, out_w, out_b, (float*)d_out);
}
